// AnalogLayer_35708358099463
// MI455X (gfx1250) — matrix operand order repaired and checked by reading, not compiled, not run on hardware
//
#include <hip/hip_runtime.h>

// ---------------------------------------------------------------------------
// Problem: P=64, M=256, B=32, BATCH=4096
//   A : [128, 512] fp32 built from p_values[64]
//   Y[131072, 128] = X[131072, 512] @ A^T
// Strategy: split-bf16 (hi/lo) 3-product WMMA emulation of fp32 GEMM.
// ---------------------------------------------------------------------------

typedef __attribute__((ext_vector_type(16))) __bf16          v16bf;
typedef __attribute__((ext_vector_type(8)))  float           v8f;
typedef __attribute__((ext_vector_type(8)))  unsigned int    v8u;

__device__ __forceinline__ unsigned short f2bf(float f) {
  unsigned u = __builtin_bit_cast(unsigned, f);
  u += 0x7FFFu + ((u >> 16) & 1u);          // round-to-nearest-even
  return (unsigned short)(u >> 16);
}
__device__ __forceinline__ float bf2f(unsigned short h) {
  return __builtin_bit_cast(float, ((unsigned)h) << 16);
}

// ---------------------------------------------------------------------------
// Build A in WMMA B-operand (32x16 K x N, bf16) register layout, hi/lo split.
// Dword index t = ((kt*8 + nt)*32 + lane)*8 + v
//   N = nt*16 + (lane & 15)
//   K = kt*32 + (lane>>4)*16 + 2*v + {0,1}   (even K in bits [15:0])
// ---------------------------------------------------------------------------
__global__ __launch_bounds__(256) void build_afrag(
    const float* __restrict__ pvals,
    unsigned* __restrict__ Ah, unsigned* __restrict__ Al) {
  const int t = blockIdx.x * blockDim.x + threadIdx.x;  // 0..32767
  const int v    = t & 7;
  const int lane = (t >> 3) & 31;
  const int nt   = (t >> 8) & 7;
  const int kt   = t >> 11;

  const int p     = nt * 16 + (lane & 15);
  const int kbase = kt * 32 + ((lane >> 4) << 4) + 2 * v;

  const float scale = -(2.0f * 3.14159265358979323846f / 256.0f);
  const float pv = pvals[p & 63];
  const int ptop = p >> 6;

  unsigned hi = 0u, lo = 0u;
#pragma unroll
  for (int j = 0; j < 2; ++j) {
    const int k    = kbase + j;
    const int mm   = k & 255;     // column within M
    const int half = k >> 8;      // 0: A1 block, 1: A2 block
    const float ang = pv * (float)mm * scale;
    const float c = cosf(ang);
    const float s = sinf(ang);
    float val;
    if (half == 0) val = ptop ? s : c;       // [cos ; sin]
    else           val = ptop ? c : -s;      // [-sin ; cos]
    const unsigned short h = f2bf(val);
    const unsigned short l = f2bf(val - bf2f(h));
    hi |= ((unsigned)h) << (16 * j);
    lo |= ((unsigned)l) << (16 * j);
  }
  Ah[t] = hi;
  Al[t] = lo;
}

// ---------------------------------------------------------------------------
// Main GEMM: 1024 blocks x 256 threads (8 waves).
// Wave w: rows [blk*128 + w*16, +16), all 128 output columns.
// Pipeline: B frags double-buffered in LDS (prefetched 2 steps ahead),
//           X fragment prefetched 1 step ahead in registers.
// ---------------------------------------------------------------------------
__global__ __launch_bounds__(256) void analog_gemm(
    const float* __restrict__ X,
    const unsigned* __restrict__ Ah,
    const unsigned* __restrict__ Al,
    float* __restrict__ Y) {
  __shared__ unsigned sh[2][4096];  // per buf: [0,2048) Bh, [2048,4096) Bl

  const int t     = threadIdx.x;
  const int lane  = t & 31;
  const int wave  = t >> 5;
  const int lrow  = lane & 15;   // M row within wave tile / N within N-tile
  const int khalf = lane >> 4;   // selects K sub-range per A-operand layout
  const int rb    = blockIdx.x * 128 + wave * 16;

  const float* __restrict__ xrow = X + (size_t)(rb + lrow) * 512;
  const v8u* __restrict__ AhV = (const v8u*)Ah;  // 4096 v8u total
  const v8u* __restrict__ AlV = (const v8u*)Al;

  v8f acc[8];
  const v8f vzero = {};
#pragma unroll
  for (int i = 0; i < 8; ++i) acc[i] = vzero;

  // --- B pipeline: fill buffer 0, prefetch step 1 into registers ---
  v8u nh = AhV[t];
  v8u nl = AlV[t];
  *(v8u*)&sh[0][t * 8]        = nh;
  *(v8u*)&sh[0][2048 + t * 8] = nl;
  nh = AhV[256 + t];
  nl = AlV[256 + t];

  // --- X pipeline: prefetch K-step 0 ---
  const int xo = khalf * 16;
  float4 c0 = *(const float4*)(xrow + xo);
  float4 c1 = *(const float4*)(xrow + xo + 4);
  float4 c2 = *(const float4*)(xrow + xo + 8);
  float4 c3 = *(const float4*)(xrow + xo + 12);

  for (int kt = 0; kt < 16; ++kt) {
    __syncthreads();  // buf (kt&1) readable; buf (kt+1)&1 free for writing

    // stage next B buffer, prefetch B two steps ahead
    if (kt < 15) {
      unsigned* nxt = sh[(kt + 1) & 1];
      *(v8u*)&nxt[t * 8]        = nh;
      *(v8u*)&nxt[2048 + t * 8] = nl;
      if (kt < 14) {
        nh = AhV[(kt + 2) * 256 + t];
        nl = AlV[(kt + 2) * 256 + t];
      }
    }

    // consume current X fragment; prefetch next K-step's X
    const float4 b0 = c0, b1 = c1, b2 = c2, b3 = c3;
    if (kt < 15) {
      const int k0 = (kt + 1) * 32 + xo;
      c0 = *(const float4*)(xrow + k0);
      c1 = *(const float4*)(xrow + k0 + 4);
      c2 = *(const float4*)(xrow + k0 + 8);
      c3 = *(const float4*)(xrow + k0 + 12);
    }

    // split fp32 -> bf16 hi/lo in 16-bit A-operand layout (native converts)
    const float f[16] = {b0.x, b0.y, b0.z, b0.w, b1.x, b1.y, b1.z, b1.w,
                         b2.x, b2.y, b2.z, b2.w, b3.x, b3.y, b3.z, b3.w};
    v16bf xh, xl;
#pragma unroll
    for (int i = 0; i < 16; ++i) {
      const __bf16 h = (__bf16)f[i];
      xh[i] = h;
      xl[i] = (__bf16)(f[i] - (float)h);
    }

    // --- 8 N-tiles x 3 split-precision WMMAs from current LDS buffer ---
    const unsigned* cur = sh[kt & 1];
#pragma unroll
    for (int nt = 0; nt < 8; ++nt) {
      const v8u bhu = *(const v8u*)&cur[(nt * 32 + lane) * 8];
      const v8u blu = *(const v8u*)&cur[2048 + (nt * 32 + lane) * 8];
      const v16bf Bh = __builtin_bit_cast(v16bf, bhu);
      const v16bf Bl = __builtin_bit_cast(v16bf, blu);
      acc[nt] = __builtin_amdgcn_wmma_f32_16x16x32_bf16(
          false, xl, false, Bh, (short)0, acc[nt], false, false);
      acc[nt] = __builtin_amdgcn_wmma_f32_16x16x32_bf16(
          false, xh, false, Bl, (short)0, acc[nt], false, false);
      acc[nt] = __builtin_amdgcn_wmma_f32_16x16x32_bf16(
          false, xh, false, Bh, (short)0, acc[nt], false, false);
    }
  }

  // --- store: D layout (16x16 f32): VGPR r -> M = khalf*8 + r, N = lrow ---
#pragma unroll
  for (int nt = 0; nt < 8; ++nt) {
    float* yp = Y + (size_t)(rb + khalf * 8) * 128 + nt * 16 + lrow;
#pragma unroll
    for (int r = 0; r < 8; ++r) {
      yp[(size_t)r * 128] = acc[nt][r];
    }
  }
}

// ---------------------------------------------------------------------------
extern "C" void kernel_launch(void* const* d_in, const int* in_sizes, int n_in,
                              void* d_out, int out_size, void* d_ws, size_t ws_size,
                              hipStream_t stream) {
  (void)in_sizes; (void)n_in; (void)out_size; (void)ws_size;

  const float* X  = (const float*)d_in[0];   // [4096, 16384] fp32
  const float* pv = (const float*)d_in[1];   // [64, 1] fp32
  float* Y = (float*)d_out;                  // [4096, 4096] fp32

  unsigned* Ah = (unsigned*)d_ws;            // 32768 dwords (128 KB)
  unsigned* Al = Ah + 32768;                 // 32768 dwords (128 KB)

  build_afrag<<<128, 256, 0, stream>>>(pv, Ah, Al);
  analog_gemm<<<1024, 256, 0, stream>>>(X, Ah, Al, Y);
}
